// DotProductAttention_24550033064222
// MI455X (gfx1250) — matrix operand order repaired and checked by reading, not compiled, not run on hardware
//
#include <hip/hip_runtime.h>
#include <hip/hip_bf16.h>

// ---------------------------------------------------------------------------
// Causal attention (B=4, S=2048, E=1024 flattened "single head") + projection
// Flash-attention structure, bf16 WMMA (v_wmma_f32_16x16x32_bf16), wave32,
// TDM (tensor_load_to_lds) double-buffered K staging, LDS-transposed V.
// Workspace layout (bytes):
//   [0,16M)   Q bf16 (pre-scaled by E^-0.5)
//   [16M,32M) K bf16
//   [32M,48M) V bf16
//   [48M,50M) W bf16 (out_w, row-major E x E)
//   [50M,66M) attn bf16 (B*S x E)
// ---------------------------------------------------------------------------

#define BB 4
#define SS 2048
#define EE 1024

// V-transpose staging: per-wave region of 128 e-rows x (32+8 pad) k-elements.
#define VT_STRIDE 40
#define VT_WAVE_ELEMS (128 * VT_STRIDE)

// K TDM staging: 32 rows x 1032 elements (1024 data + 8 pad) = 516 dwords/row.
// 516 % 64 == 4 -> row-lanes of b128 reads hit distinct bank groups; row
// stride 2064B stays 16B-aligned.
#define KROW_ELEMS 1032
#define KROW_DW    516
#define KBUF_DW    (32 * KROW_DW)          // 16512 dwords = 66048 B
#define FA_DYN_LDS (2 * KBUF_DW * 4 + 8 * VT_WAVE_ELEMS * 2)  // 214016 B

typedef __attribute__((ext_vector_type(16))) __bf16 v16bf;
typedef __attribute__((ext_vector_type(4)))  __bf16 v4bf;
typedef __attribute__((ext_vector_type(8)))  float  v8f;
typedef __attribute__((ext_vector_type(4)))  unsigned int tdm_g0_t;
typedef __attribute__((ext_vector_type(8)))  int          tdm_g1_t;
typedef __attribute__((ext_vector_type(4)))  int          tdm_g2_t;

// Device-side TDM availability (host pass sees 0; host only sizes LDS).
#if defined(__AMDGCN__) && defined(__has_builtin)
#if __has_builtin(__builtin_amdgcn_tensor_load_to_lds) && \
    __has_builtin(__builtin_amdgcn_s_wait_tensorcnt)
#define USE_TDM 1
#endif
#endif
#ifndef USE_TDM
#define USE_TDM 0
#endif

#if USE_TDM
// Issue one TDM load: 32 K-rows of 1024 bf16, row-strided into LDS.
// D# per CDNA5 ISA ch.8: group0 {count, lds_addr, global_addr, type=2},
// group1 {data_size=2B | iterate_enable, tensor/tile dims, dim0 stride},
// group2 {iterate: lds_addr_increment, global_addr_increment, count}.
__device__ __forceinline__ void tdm_load_k_block(const __bf16* gsrc,
                                                 unsigned lds_byte) {
  const unsigned long long ga = (unsigned long long)(const void*)gsrc;
  tdm_g0_t g0 = { 1u,                                   // count=1
                  lds_byte,                             // lds_addr (bytes)
                  (unsigned)ga,                         // global_addr[31:0]
                  (unsigned)((ga >> 32) & 0x01FFFFFFull) | (2u << 30) };
  tdm_g1_t g1 = { (int)0x00090000,       // data_size=2B (1<<16) | iterate (1<<19)
                  (int)(1024 << 16),     // tensor_dim0 = 1024 (bits 63:48)
                  (int)(1 << 16),        // tensor_dim1 = 1    (bits 95:80)
                  (int)(1024 << 16),     // tile_dim0  = 1024  (bits 127:112)
                  1,                     // tile_dim1 = 1, tile_dim2 = 0
                  1024,                  // tensor_dim0_stride (elements)
                  0, 0 };
  tdm_g2_t g2 = { 0,                     // tensor_dim2
                  KROW_ELEMS,            // lds_addr_increment (elements)
                  1024,                  // global_addr_increment (elements)
                  (int)(31 << 16) };     // iterate_count = 31 -> 32 rows
  tdm_g2_t g3 = { 0, 0, 0, 0 };
#if __clang_major__ >= 23
  tdm_g1_t g4 = { 0, 0, 0, 0, 0, 0, 0, 0 };
  __builtin_amdgcn_tensor_load_to_lds(g0, g1, g2, g3, g4, 0);
#else
  __builtin_amdgcn_tensor_load_to_lds(g0, g1, g2, g3, 0);
#endif
}
#endif

// ----------------------------- f32 -> bf16 ---------------------------------
__global__ void cvt_bf16_4(const float* __restrict__ src,
                           __bf16* __restrict__ dst,
                           long n4, float scale) {
  long i = (long)blockIdx.x * blockDim.x + threadIdx.x;
  long stride = (long)gridDim.x * blockDim.x;
  for (; i < n4; i += stride) {
    float4 f = ((const float4*)src)[i];
    v4bf h;
    h.x = (__bf16)(f.x * scale);
    h.y = (__bf16)(f.y * scale);
    h.z = (__bf16)(f.z * scale);
    h.w = (__bf16)(f.w * scale);
    *(v4bf*)(dst + 4 * i) = h;
  }
}

// --------------------------- flash attention -------------------------------
// Grid: B*S/16 workgroups, 256 threads (8 waves). Wave w owns E-slice
// [w*128, w*128+128). Cross-wave score reduction in LDS via ds_add_f32.
// Dynamic LDS: [2 x K TDM buffer][8 x V-transpose region].
__launch_bounds__(256, 1)
__global__ void flash_attn(const __bf16* __restrict__ qb,
                           const __bf16* __restrict__ kb,
                           const __bf16* __restrict__ vb,
                           __bf16* __restrict__ attn) {
  __shared__ float sScore[16 * 32];  // [query row][key in block]
  __shared__ float sMnew[16];
  __shared__ float sFactor[16];
  __shared__ float sLinv[16];
  extern __shared__ unsigned int dynLDS[];

  unsigned int*   kbuf0 = dynLDS;                 // KBUF_DW dwords
  unsigned int*   kbuf1 = dynLDS + KBUF_DW;
  unsigned short* sVT   = (unsigned short*)(dynLDS + 2 * KBUF_DW);

  const int wg    = blockIdx.x;
  const int batch = wg / (SS / 16);
  const int qbase = (wg % (SS / 16)) * 16;
  const int lane  = threadIdx.x & 31;
  const int wave  = threadIdx.x >> 5;
  const int half  = lane >> 4;   // 0 / 1 (lane group)
  const int l15   = lane & 15;
  const int eslc  = wave * 128;

  unsigned short* vt = sVT + wave * VT_WAVE_ELEMS;  // wave-private region

  const __bf16* Qb = qb + (size_t)batch * SS * EE;
  const __bf16* Kb = kb + (size_t)batch * SS * EE;
  const __bf16* Vb = vb + (size_t)batch * SS * EE;

  // Q A-operands: 4 chunks of 16x32 bf16 (two contiguous 16B runs per lane).
  v16bf Aq[4];
#pragma unroll
  for (int c = 0; c < 4; ++c) {
    const __bf16* p = Qb + (size_t)(qbase + l15) * EE + (eslc + c * 32 + half * 16);
    union { uint4 u[2]; v16bf v; } t;
    t.u[0] = *(const uint4*)p;
    t.u[1] = *(const uint4*)(p + 8);
    Aq[c] = t.v;
  }

  v8f acc[8] = {};          // 8 output tiles of 16x16 f32 (this wave's 128 cols)
  float m_run = -3.0e38f;   // wave 0, lanes 0..15: row stats (row = lane)
  float l_run = 0.0f;

  const int qlast = qbase + 15;
  const int nblk  = qlast / 32 + 1;
  const float L2E = 1.44269504f;

#if USE_TDM
  // Pre-issue TDM loads for key blocks 0 (and 1) into the double buffer.
  if (threadIdx.x < 32) {
    tdm_load_k_block(Kb, (unsigned)(unsigned long long)(const void*)kbuf0);
    if (nblk > 1)
      tdm_load_k_block(Kb + (size_t)32 * EE,
                       (unsigned)(unsigned long long)(const void*)kbuf1);
  }
#endif

  for (int blk = 0; blk < nblk; ++blk) {
    const int kbase = blk * 32;

    // zero score reduction buffer (512 floats / 256 threads)
    sScore[threadIdx.x]       = 0.0f;
    sScore[threadIdx.x + 256] = 0.0f;

    // prefetch next block's V rows for this wave's E-slice
    if (blk + 1 < nblk) {
      const int pkrow = kbase + 32 + l15 + (half ? 16 : 0);
      __builtin_prefetch(Vb + (size_t)pkrow * EE + eslc, 0, 0);
#if !USE_TDM
      __builtin_prefetch(Kb + (size_t)pkrow * EE + eslc, 0, 0);
#endif
    }

    // ---- stage this wave's V slice (32 keys x 128 e) transposed to LDS --
#pragma unroll
    for (int j = 0; j < 8; ++j) {
      const int eg = j * 2 + half;    // e-group of 8 (0..15)
      const int e0 = eg * 8;
      const int k0 = 2 * l15;         // adjacent key rows k0, k0+1
      const __bf16* p0 = Vb + (size_t)(kbase + k0) * EE + eslc + e0;
      union { uint4 u; unsigned short s[8]; } ra, rb;
      ra.u = *(const uint4*)p0;
      rb.u = *(const uint4*)(p0 + EE);
#pragma unroll
      for (int i = 0; i < 8; ++i) {
        const unsigned int d =
            (unsigned int)ra.s[i] | ((unsigned int)rb.s[i] << 16);
        *(unsigned int*)&vt[(e0 + i) * VT_STRIDE + k0] = d;
      }
    }

#if USE_TDM
    // Wave 0 issued the TDM loads; make the current K buffer visible.
    // With a younger load possibly in flight, oldest-done needs cnt<=1;
    // on the last block only the current load is outstanding -> cnt<=0.
    if (threadIdx.x < 32) {
      if (blk + 1 < nblk) __builtin_amdgcn_s_wait_tensorcnt(1);
      else                __builtin_amdgcn_s_wait_tensorcnt(0);
    }
#endif
    __syncthreads();

    // ---- QK^T partials over this wave's 128-dim slice -------------------
    // Preload all 8 B-tiles into distinct registers so the 16 ds_load_b128
    // stay in flight and the WMMA burst amortizes s_wait_dscnt.
    union BU { uint4 u[2]; v16bf v; };
    BU b0[4], b1[4];
#if USE_TDM
    const unsigned int* kcur = (blk & 1) ? kbuf1 : kbuf0;
#pragma unroll
    for (int c = 0; c < 4; ++c) {
      const int dw = (eslc + c * 32 + half * 16) >> 1;
      const uint4* p0 = (const uint4*)(kcur + l15 * KROW_DW + dw);
      const uint4* p1 = (const uint4*)(kcur + (l15 + 16) * KROW_DW + dw);
      b0[c].u[0] = p0[0]; b0[c].u[1] = p0[1];
      b1[c].u[0] = p1[0]; b1[c].u[1] = p1[1];
    }
#else
#pragma unroll
    for (int c = 0; c < 4; ++c) {
      const int e0 = eslc + c * 32 + half * 16;
      const __bf16* pk0 = Kb + (size_t)(kbase + l15) * EE + e0;
      const __bf16* pk1 = Kb + (size_t)(kbase + 16 + l15) * EE + e0;
      b0[c].u[0] = *(const uint4*)pk0; b0[c].u[1] = *(const uint4*)(pk0 + 8);
      b1[c].u[0] = *(const uint4*)pk1; b1[c].u[1] = *(const uint4*)(pk1 + 8);
    }
#endif
    v8f s0 = {}, s1 = {};
#pragma unroll
    for (int c = 0; c < 4; ++c) {
      s0 = __builtin_amdgcn_wmma_f32_16x16x32_bf16(false, Aq[c], false, b0[c].v,
                                                   (short)0, s0, false, false);
      s1 = __builtin_amdgcn_wmma_f32_16x16x32_bf16(false, Aq[c], false, b1[c].v,
                                                   (short)0, s1, false, false);
    }

    // ---- cross-wave reduction (C-layout: row = r + 8*half, col = l15) ---
#pragma unroll
    for (int r = 0; r < 8; ++r) {
      const int row = r + 8 * half;
      atomicAdd(&sScore[row * 32 + l15], s0[r]);
      atomicAdd(&sScore[row * 32 + 16 + l15], s1[r]);
    }
    __syncthreads();

    // ---- online softmax stats (wave 0, one lane per query row) ----------
    if (wave == 0 && lane < 16) {
      const int q = qbase + lane;
      float mx = m_run;
#pragma unroll 8
      for (int j = 0; j < 32; ++j) {
        const float s = sScore[lane * 32 + j];
        if ((kbase + j) <= q && s > mx) mx = s;
      }
      const float factor = __builtin_amdgcn_exp2f((m_run - mx) * L2E);
      float sum = 0.0f;
#pragma unroll 8
      for (int j = 0; j < 32; ++j) {
        const float s = sScore[lane * 32 + j];
        sum += ((kbase + j) <= q)
                   ? __builtin_amdgcn_exp2f((s - mx) * L2E) : 0.0f;
      }
      l_run = l_run * factor + sum;
      m_run = mx;
      sMnew[lane]   = mx;
      sFactor[lane] = factor;
    }
    __syncthreads();

    // ---- build P (16x32 bf16, A-layout) and rescale accumulators --------
    const float mnew = sMnew[l15];
    const int   q    = qbase + l15;
    union { v16bf v; __bf16 h[16]; } P;
#pragma unroll
    for (int i = 0; i < 16; ++i) {
      const int kk = (half << 4) + i;
      const float s = sScore[l15 * 32 + kk];
      const float p = ((kbase + kk) <= q)
                          ? __builtin_amdgcn_exp2f((s - mnew) * L2E) : 0.0f;
      P.h[i] = (__bf16)p;
    }
#pragma unroll
    for (int r = 0; r < 8; ++r) {
      const float f = sFactor[r + 8 * half];
#pragma unroll
      for (int t = 0; t < 8; ++t) acc[t][r] *= f;
    }

    // ---- PV: acc[t] += P (16x32) x V^T tile from LDS --------------------
#pragma unroll
    for (int t = 0; t < 8; ++t) {
      const int ep = 16 * t + l15;
      const unsigned short* bp = &vt[ep * VT_STRIDE + 16 * half];
      union { uint4 u[2]; v16bf v; } Bv;
      Bv.u[0] = *(const uint4*)bp;
      Bv.u[1] = *(const uint4*)(bp + 8);
      acc[t] = __builtin_amdgcn_wmma_f32_16x16x32_bf16(false, P.v, false, Bv.v,
                                                       (short)0, acc[t], false, false);
    }
    __syncthreads();  // all waves done with sScore and current K buffer

#if USE_TDM
    // Refill the buffer just released with key block blk+2.
    if (threadIdx.x < 32 && (blk + 2) < nblk) {
      unsigned int* nb = (blk & 1) ? kbuf1 : kbuf0;
      tdm_load_k_block(Kb + (size_t)(blk + 2) * 32 * EE,
                       (unsigned)(unsigned long long)(const void*)nb);
    }
#endif
  }

  // ---- finalize: divide by row sums, store bf16 attn tile ---------------
  if (wave == 0 && lane < 16) sLinv[lane] = 1.0f / l_run;
  __syncthreads();
#pragma unroll
  for (int t = 0; t < 8; ++t) {
    const int e = eslc + 16 * t + l15;
#pragma unroll
    for (int r = 0; r < 8; ++r) {
      const int row = r + 8 * half;
      const float o = acc[t][r] * sLinv[row];
      attn[(size_t)(batch * SS + qbase + row) * EE + e] = (__bf16)o;
    }
  }
}

// ------------------------- output projection GEMM --------------------------
// out[M=B*S][N=E] = attn[M][K=E] * W[N][K]^T + bias ; f32 output.
__launch_bounds__(256, 1)
__global__ void proj_gemm(const __bf16* __restrict__ A,
                          const __bf16* __restrict__ W,
                          const float* __restrict__ bias,
                          float* __restrict__ out) {
  const int tileN = (blockIdx.x & 7) * 128;
  const int tileM = (blockIdx.x >> 3) * 128;
  const int lane  = threadIdx.x & 31;
  const int wave  = threadIdx.x >> 5;
  const int half  = lane >> 4;
  const int l15   = lane & 15;
  const int row0  = tileM + wave * 16;

  v8f acc[8] = {};

  for (int kk = 0; kk < EE; kk += 32) {
    union { uint4 u[2]; v16bf v; } Av;
    const __bf16* pa = A + (size_t)(row0 + l15) * EE + kk + half * 16;
    Av.u[0] = *(const uint4*)pa;
    Av.u[1] = *(const uint4*)(pa + 8);
#pragma unroll
    for (int t = 0; t < 8; ++t) {
      const __bf16* pw = W + (size_t)(tileN + 16 * t + l15) * EE + kk + half * 16;
      union { uint4 u[2]; v16bf v; } Bv;
      Bv.u[0] = *(const uint4*)pw;
      Bv.u[1] = *(const uint4*)(pw + 8);
      acc[t] = __builtin_amdgcn_wmma_f32_16x16x32_bf16(false, Av.v, false, Bv.v,
                                                       (short)0, acc[t], false, false);
    }
  }
#pragma unroll
  for (int t = 0; t < 8; ++t) {
    const int n = tileN + 16 * t + l15;
    const float bv = bias[n];
#pragma unroll
    for (int r = 0; r < 8; ++r) {
      const int m = row0 + r + 8 * half;
      out[(size_t)m * EE + n] = acc[t][r] + bv;
    }
  }
}

// ------------------------------- launcher ----------------------------------
extern "C" void kernel_launch(void* const* d_in, const int* in_sizes, int n_in,
                              void* d_out, int out_size, void* d_ws, size_t ws_size,
                              hipStream_t stream) {
  (void)in_sizes; (void)n_in; (void)out_size; (void)ws_size;

  const float* q = (const float*)d_in[0];
  const float* k = (const float*)d_in[1];
  const float* v = (const float*)d_in[2];
  // d_in[3] qkv_proj: unused. d_in[4] attn_mask: causal, applied analytically.
  const float* w = (const float*)d_in[5];
  const float* b = (const float*)d_in[6];
  float* out = (float*)d_out;

  const size_t N = (size_t)BB * SS * EE;  // 8,388,608
  char* ws = (char*)d_ws;
  __bf16* qb   = (__bf16*)(ws);
  __bf16* kb   = (__bf16*)(ws + 2 * N);
  __bf16* vb   = (__bf16*)(ws + 4 * N);
  __bf16* wb   = (__bf16*)(ws + 6 * N);
  __bf16* attn = (__bf16*)(ws + 6 * N + 2 * (size_t)EE * EE);

  const float qscale = 1.0f / 32.0f;  // E^-0.5, E = 1024

  cvt_bf16_4<<<2048, 256, 0, stream>>>(q, qb, (long)(N / 4), qscale);
  cvt_bf16_4<<<2048, 256, 0, stream>>>(k, kb, (long)(N / 4), 1.0f);
  cvt_bf16_4<<<2048, 256, 0, stream>>>(v, vb, (long)(N / 4), 1.0f);
  cvt_bf16_4<<<512, 256, 0, stream>>>(w, wb, (long)((size_t)EE * EE / 4), 1.0f);

  // Host always sizes for the TDM layout (device may or may not use it).
  flash_attn<<<BB * (SS / 16), 256, FA_DYN_LDS, stream>>>(qb, kb, vb, attn);

  const int gM = (BB * SS) / 128;  // 64
  const int gN = EE / 128;         // 8
  proj_gemm<<<gM * gN, 256, 0, stream>>>(attn, wb, b, out);
}
